// ActorBatchNet_64467459113215
// MI455X (gfx1250) — matrix operand order repaired and checked by reading, not compiled, not run on hardware
//
#include <hip/hip_runtime.h>
#include <hip/hip_bf16.h>

typedef __attribute__((ext_vector_type(16))) _Float16 v16h;
typedef __attribute__((ext_vector_type(8)))  _Float16 v8h;
typedef __attribute__((ext_vector_type(8)))  float    v8f;

#define DIMH   64
#define NNODE  20480
#define NEDGE  43008
#define NGRAPH 512
#define NACT   6
#define NTOR   3072
#define KBMSG  130          // 128 blocks of K=32 (i,k pairs) + 2 blocks for enn2 bias rows

__device__ __forceinline__ float sigmf(float x) { return 1.0f / (1.0f + __expf(-x)); }

// ---------------- simple elementwise / setup kernels ----------------

__global__ void k_zero(float* p, int n) {
    int i = blockIdx.x * 256 + threadIdx.x;
    if (i < n) p[i] = 0.0f;
}

__global__ void k_lin0(const float* __restrict__ x, const float* __restrict__ w,
                       const float* __restrict__ b, float* __restrict__ h) {
    int idx = blockIdx.x * 256 + threadIdx.x;
    if (idx >= NNODE * DIMH) return;
    int n = idx >> 6, d = idx & 63;
    float acc = b[d] + x[n * 3 + 0] * w[d * 3 + 0] + x[n * 3 + 1] * w[d * 3 + 1]
              + x[n * 3 + 2] * w[d * 3 + 2];
    h[idx] = fmaxf(acc, 0.0f);
}

__global__ void k_enn1(const float* __restrict__ ea, const float* __restrict__ w,
                       const float* __restrict__ b, float* __restrict__ ew) {
    int idx = blockIdx.x * 256 + threadIdx.x;
    if (idx >= NEDGE * DIMH) return;
    int e = idx >> 6, d = idx & 63;
    float acc = b[d];
#pragma unroll
    for (int k = 0; k < 6; ++k) acc += ea[e * 6 + k] * w[d * 6 + k];
    ew[idx] = fmaxf(acc, 0.0f);
}

__global__ void k_deg(const int* __restrict__ dst, float* __restrict__ deg) {
    int e = blockIdx.x * 256 + threadIdx.x;
    if (e < NEDGE) atomicAdd(&deg[dst[e]], 1.0f);
}

__global__ void k_invdeg(float* d) {
    int n = blockIdx.x * 256 + threadIdx.x;
    if (n < NNODE) d[n] = 1.0f / fmaxf(d[n], 1.0f);
}

// ---------------- B-matrix fragment packing (ISA B 32x16 f16 layout) ----------------
// Bp[((kb*nblk + nt)*32 + lane)*16 + t]  holds  G[kb*32 + 8*(lane>=16) + (t&7) + 16*(t>>3)][nt*16 + (lane&15)]

__global__ void k_pack_msg(const float* __restrict__ w2, const float* __restrict__ b2,
                           _Float16* __restrict__ bp) {
    int idx = blockIdx.x * 256 + threadIdx.x;
    if (idx >= KBMSG * 4 * 512) return;
    int t = idx & 15, lane = (idx >> 4) & 31, nt = (idx >> 9) & 3, kb = idx >> 11;
    int k = kb * 32 + ((lane >> 4) << 3) + (t & 7) + ((t >> 3) << 4);
    int o = nt * 16 + (lane & 15);
    float v;
    if (k < 4096) v = w2[(size_t)((k >> 6) * 64 + o) * 64 + (k & 63)];   // enn2_w[(i*64+o), kk]
    else          v = b2[(k - 4096) * 64 + o];                           // bias rows
    bp[idx] = (_Float16)v;
}

// G given transposed: math G[k][f] = wt[f*K + k]   (for  X @ W^T  style layers)
__global__ void k_pack_T(const float* __restrict__ wt, _Float16* __restrict__ bp, int K, int F) {
    int total = (K >> 5) * (F >> 4) * 512;
    int idx = blockIdx.x * 256 + threadIdx.x;
    if (idx >= total) return;
    int t = idx & 15, lane = (idx >> 4) & 31, rest = idx >> 9;
    int nblk = F >> 4;
    int nt = rest % nblk, kb = rest / nblk;
    int k = kb * 32 + ((lane >> 4) << 3) + (t & 7) + ((t >> 3) << 4);
    int f = nt * 16 + (lane & 15);
    bp[idx] = (_Float16)wt[(size_t)f * K + k];
}

// G row-major (K,F): math G[k][f] = g[k*F + f]   (for  X @ W  style, e.g. root_w)
__global__ void k_pack_G(const float* __restrict__ g, _Float16* __restrict__ bp, int K, int F) {
    int total = (K >> 5) * (F >> 4) * 512;
    int idx = blockIdx.x * 256 + threadIdx.x;
    if (idx >= total) return;
    int t = idx & 15, lane = (idx >> 4) & 31, rest = idx >> 9;
    int nblk = F >> 4;
    int nt = rest % nblk, kb = rest / nblk;
    int k = kb * 32 + ((lane >> 4) << 3) + (t & 7) + ((t >> 3) << 4);
    int f = nt * 16 + (lane & 15);
    bp[idx] = (_Float16)g[(size_t)k * F + f];
}

// ---------------- generic WMMA GEMM: C(MxF) = A(MxK) @ Bp + bias ----------------
// grid(M/16, F/64), block(128) = 4 waves; wave w owns n-tile blockIdx.y*4+w.
// K unrolled by 2 with independent accumulators (K/32 is even in all uses).
// Optional fused NNConv epilogue: C = relu(acc + bias + aggadd*invd[row]).

__global__ __launch_bounds__(128)
void k_gemm_wmma(const float* __restrict__ A, const _Float16* __restrict__ Bp,
                 const float* __restrict__ bias, float* __restrict__ C,
                 int K, int F, int do_relu,
                 const float* __restrict__ aggadd, const float* __restrict__ invd) {
    int mt = blockIdx.x;
    int nt = blockIdx.y * 4 + (threadIdx.x >> 5);
    int l  = threadIdx.x & 31;
    int m  = mt * 16 + (l & 15);
    int hi = (l >> 4) & 1;
    int nblk = F >> 4;
    const _Float16* bbase = Bp + ((size_t)nt * 32 + l) * 16;
    __builtin_prefetch((const void*)bbase, 0, 3);
    v8f acc0 = {}, acc1 = {};
    int KB = K >> 5;
    for (int kb = 0; kb < KB; kb += 2) {
        v16h a0, a1, b0, b1;
        const float* arow0 = A + (size_t)m * K + (kb << 5) + (hi << 3);
        const float* arow1 = arow0 + 32;
#pragma unroll
        for (int t = 0; t < 8; ++t) a0[t]     = (_Float16)arow0[t];       // K = 8*hi + t
#pragma unroll
        for (int t = 0; t < 8; ++t) a0[8 + t] = (_Float16)arow0[16 + t];  // K = 8*hi + 16 + t
#pragma unroll
        for (int t = 0; t < 8; ++t) a1[t]     = (_Float16)arow1[t];
#pragma unroll
        for (int t = 0; t < 8; ++t) a1[8 + t] = (_Float16)arow1[16 + t];
        const _Float16* brow0 = bbase + (size_t)kb * nblk * 512;
        const _Float16* brow1 = brow0 + (size_t)nblk * 512;
#pragma unroll
        for (int t = 0; t < 16; ++t) b0[t] = brow0[t];
#pragma unroll
        for (int t = 0; t < 16; ++t) b1[t] = brow1[t];
        acc0 = __builtin_amdgcn_wmma_f32_16x16x32_f16(false, a0, false, b0, (short)0, acc0,
                                                      false, false);
        acc1 = __builtin_amdgcn_wmma_f32_16x16x32_f16(false, a1, false, b1, (short)0, acc1,
                                                      false, false);
    }
    v8f acc = acc0 + acc1;
    int col = nt * 16 + (l & 15);
    float bv = bias ? bias[col] : 0.0f;
#pragma unroll
    for (int r = 0; r < 8; ++r) {
        int row = mt * 16 + r + (hi << 3);
        float v = acc[r] + bv;
        if (aggadd) v += aggadd[(size_t)row * F + col] * invd[row];
        if (do_relu) v = fmaxf(v, 0.0f);
        C[(size_t)row * F + col] = v;
    }
}

// ---------------- fused edge-message GEMM + scatter-add ----------------
// msg_e = outer(out[src_e], ew_e)(4096) @ W2r(4160x64)  -> atomicAdd into agg[dst_e]
// A-frag generated in-regs via packed f16 math from f16 LDS tiles.
// Loop over 65 k-block pairs; each pair shares one xs scalar and covers a full ew row.

__global__ __launch_bounds__(128)
void k_msg_wmma(const float* __restrict__ H, const float* __restrict__ EW,
                const _Float16* __restrict__ Bp, const int* __restrict__ src,
                const int* __restrict__ dst, float* __restrict__ agg) {
    __shared__ alignas(16) _Float16 xs_h[16][64];
    __shared__ alignas(16) _Float16 ew_h[16][64];
    __shared__ int dsts[16];
    int e0 = blockIdx.x * 16;
    int tid = threadIdx.x;
    for (int idx = tid; idx < 1024; idx += 128) {
        int row = idx >> 6, col = idx & 63;
        int e = e0 + row;
        xs_h[row][col] = (_Float16)H[(size_t)src[e] * 64 + col];
        ew_h[row][col] = (_Float16)EW[(size_t)e * 64 + col];
    }
    if (tid < 16) dsts[tid] = dst[e0 + tid];
    __syncthreads();

    int w  = tid >> 5;       // wave id == n-tile
    int l  = tid & 31;
    int m  = l & 15;
    int hi = (l >> 4) & 1;
    const _Float16* bbase = Bp + ((size_t)w * 32 + l) * 16;
    __builtin_prefetch((const void*)bbase, 0, 3);
    v8f acc0 = {}, acc1 = {};
    int hb = hi << 3;
    for (int q = 0; q < KBMSG / 2; ++q) {          // pair (kb=2q, kb=2q+1)
        v16h a0, a1, b0, b1;
        if (q < 64) {
            v8h e00 = *(const v8h*)&ew_h[m][hb];           // kk: hb..hb+7
            v8h e01 = *(const v8h*)&ew_h[m][hb + 16];      // kk: hb+16..hb+23
            v8h e10 = *(const v8h*)&ew_h[m][hb + 32];      // kk: hb+32..
            v8h e11 = *(const v8h*)&ew_h[m][hb + 48];
            _Float16 xvh = xs_h[m][q];
            v8h xsp;
#pragma unroll
            for (int t = 0; t < 8; ++t) xsp[t] = xvh;
            e00 = e00 * xsp; e01 = e01 * xsp;              // v_pk_mul_f16
            e10 = e10 * xsp; e11 = e11 * xsp;
#pragma unroll
            for (int t = 0; t < 8; ++t) { a0[t] = e00[t]; a0[8 + t] = e01[t]; }
#pragma unroll
            for (int t = 0; t < 8; ++t) { a1[t] = e10[t]; a1[8 + t] = e11[t]; }
        } else {                                           // bias rows: P[4096+j] = xs[j]
            v8h x00 = *(const v8h*)&xs_h[m][hb];
            v8h x01 = *(const v8h*)&xs_h[m][hb + 16];
            v8h x10 = *(const v8h*)&xs_h[m][hb + 32];
            v8h x11 = *(const v8h*)&xs_h[m][hb + 48];
#pragma unroll
            for (int t = 0; t < 8; ++t) { a0[t] = x00[t]; a0[8 + t] = x01[t]; }
#pragma unroll
            for (int t = 0; t < 8; ++t) { a1[t] = x10[t]; a1[8 + t] = x11[t]; }
        }
        const _Float16* brow0 = bbase + (size_t)(2 * q) * 4 * 512;
        const _Float16* brow1 = brow0 + 4 * 512;
#pragma unroll
        for (int t = 0; t < 16; ++t) b0[t] = brow0[t];
#pragma unroll
        for (int t = 0; t < 16; ++t) b1[t] = brow1[t];
        acc0 = __builtin_amdgcn_wmma_f32_16x16x32_f16(false, a0, false, b0, (short)0, acc0,
                                                      false, false);
        acc1 = __builtin_amdgcn_wmma_f32_16x16x32_f16(false, a1, false, b1, (short)0, acc1,
                                                      false, false);
    }
    v8f acc = acc0 + acc1;
    int o = w * 16 + m;
#pragma unroll
    for (int r = 0; r < 8; ++r) {
        int erow = r + hb;
        atomicAdd(&agg[(size_t)dsts[erow] * 64 + o], acc[r]);
    }
}

// ---------------- GRU combine ----------------

__global__ void k_gru(const float* __restrict__ gi, const float* __restrict__ gh,
                      float* __restrict__ h) {
    int idx = blockIdx.x * 256 + threadIdx.x;
    if (idx >= NNODE * DIMH) return;
    int n = idx >> 6, j = idx & 63;
    const float* gin = gi + (size_t)n * 192;
    const float* ghn = gh + (size_t)n * 192;
    float r  = sigmf(gin[j] + ghn[j]);
    float z  = sigmf(gin[64 + j] + ghn[64 + j]);
    float nn = tanhf(gin[128 + j] + r * ghn[128 + j]);
    h[idx] = (1.0f - z) * nn + z * h[idx];
}

// ---------------- LSTM cell (Set2Set + memory), one block per graph ----------------

__global__ __launch_bounds__(64)
void k_lstm(const float* __restrict__ xin, int in_dim,
            const float* __restrict__ w_ih, const float* __restrict__ b_ih,
            const float* __restrict__ w_hh, const float* __restrict__ b_hh,
            float* __restrict__ hbuf, float* __restrict__ cbuf, float* __restrict__ qout) {
    int b = blockIdx.x, j = threadIdx.x;
    const float* x  = xin + (size_t)b * in_dim;
    const float* hp = hbuf + (size_t)b * 64;
    float g[4];
#pragma unroll
    for (int gg = 0; gg < 4; ++gg) {
        int row = gg * 64 + j;
        float acc = b_ih[row] + b_hh[row];
        const float* wi = w_ih + (size_t)row * in_dim;
        for (int k = 0; k < in_dim; ++k) acc += x[k] * wi[k];
        const float* wh = w_hh + (size_t)row * 64;
        for (int k = 0; k < 64; ++k) acc += hp[k] * wh[k];
        g[gg] = acc;
    }
    float c  = cbuf[(size_t)b * 64 + j];
    float c2 = sigmf(g[1]) * c + sigmf(g[0]) * tanhf(g[2]);
    float h2 = sigmf(g[3]) * tanhf(c2);
    __syncthreads();                    // all reads of hbuf row done before writes
    hbuf[(size_t)b * 64 + j] = h2;
    cbuf[(size_t)b * 64 + j] = c2;
    qout[(size_t)b * 64 + j] = h2;
}

// ---------------- Set2Set attention over 40 contiguous nodes per graph ----------------

__global__ __launch_bounds__(64)
void k_attn(const float* __restrict__ h, const float* __restrict__ q,
            float* __restrict__ qstar) {
    __shared__ float ebuf[40];
    __shared__ float qs[64];
    __shared__ float sinv;
    int b = blockIdx.x, j = threadIdx.x;
    qs[j] = q[(size_t)b * 64 + j];
    __syncthreads();
    if (j < 40) {
        const float* hn = h + ((size_t)b * 40 + j) * 64;
        float e = 0.0f;
        for (int d = 0; d < 64; ++d) e += hn[d] * qs[d];
        ebuf[j] = e;
    }
    __syncthreads();
    if (j == 0) {
        float mx = ebuf[0];
        for (int n = 1; n < 40; ++n) mx = fmaxf(mx, ebuf[n]);
        float s = 0.0f;
        for (int n = 0; n < 40; ++n) { ebuf[n] = __expf(ebuf[n] - mx); s += ebuf[n]; }
        sinv = 1.0f / s;
    }
    __syncthreads();
    float r = 0.0f;
    for (int n = 0; n < 40; ++n) r += ebuf[n] * h[((size_t)b * 40 + n) * 64 + j];
    r *= sinv;
    qstar[(size_t)b * 128 + j]      = qs[j];
    qstar[(size_t)b * 128 + 64 + j] = r;
}

// ---------------- final head ----------------

// z2[row][col]: row = b*6+q, flat within b: u = q*320+col = d*30 + a*5 + f
__global__ void k_z2(const float* __restrict__ lstm_out, const float* __restrict__ h,
                     const int* __restrict__ nonring, float* __restrict__ z2) {
    int idx = blockIdx.x * 256 + threadIdx.x;
    if (idx >= NTOR * 320) return;
    int row = idx / 320, col = idx - row * 320;
    int b = row / 6, q = row - b * 6;
    int u = q * 320 + col;
    int d = u / 30;
    int rem = u - d * 30;
    int a = rem / 5, f = rem - a * 5;
    float v;
    if (f == 0) v = lstm_out[(size_t)b * 64 + d];
    else        v = h[(size_t)nonring[(b * 6 + a) * 4 + (f - 1)] * 64 + d];
    z2[idx] = v;
}

__global__ void k_head(const float* __restrict__ z2h, const float* __restrict__ w,
                       const float* __restrict__ bias, float* __restrict__ out) {
    int idx = blockIdx.x * 256 + threadIdx.x;
    if (idx >= NTOR * 6) return;
    int t = idx / 6, j = idx - t * 6;
    float acc = bias[j];
    const float* zr = z2h + (size_t)t * 64;
    for (int d = 0; d < 64; ++d) acc += zr[d] * w[j * 64 + d];
    out[idx] = acc;
}

// ---------------- host launcher ----------------

static inline int cdiv(int a, int b) { return (a + b - 1) / b; }

extern "C" void kernel_launch(void* const* d_in, const int* in_sizes, int n_in,
                              void* d_out, int out_size, void* d_ws, size_t ws_size,
                              hipStream_t stream) {
    const float* x         = (const float*)d_in[0];
    const float* edge_attr = (const float*)d_in[1];
    const int*   eidx      = (const int*)d_in[2];
    const int*   nonring   = (const int*)d_in[5];
    const float* lin0_w = (const float*)d_in[6];  const float* lin0_b = (const float*)d_in[7];
    const float* enn1_w = (const float*)d_in[8];  const float* enn1_b = (const float*)d_in[9];
    const float* enn2_w = (const float*)d_in[10]; const float* enn2_b = (const float*)d_in[11];
    const float* root_w = (const float*)d_in[12]; const float* conv_b = (const float*)d_in[13];
    const float* gru_w_ih = (const float*)d_in[14]; const float* gru_b_ih = (const float*)d_in[15];
    const float* gru_w_hh = (const float*)d_in[16]; const float* gru_b_hh = (const float*)d_in[17];
    const float* s2s_w_ih = (const float*)d_in[18]; const float* s2s_b_ih = (const float*)d_in[19];
    const float* s2s_w_hh = (const float*)d_in[20]; const float* s2s_b_hh = (const float*)d_in[21];
    const float* mem_w_ih = (const float*)d_in[22]; const float* mem_b_ih = (const float*)d_in[23];
    const float* mem_w_hh = (const float*)d_in[24]; const float* mem_b_hh = (const float*)d_in[25];
    const float* lin1_w = (const float*)d_in[26]; const float* lin1_b = (const float*)d_in[27];
    const float* lin2_w = (const float*)d_in[28]; const float* lin2_b = (const float*)d_in[29];

    const int* src = eidx;
    const int* dst = eidx + NEDGE;

    float* out = (float*)d_out;
    float* hx  = out + NGRAPH * NACT * NACT;   // +18432
    float* cx  = hx + NGRAPH * DIMH;           // +32768

    // workspace carve (f32)
    float* p = (float*)d_ws;
    float* h     = p; p += NNODE * DIMH;
    float* agg   = p; p += NNODE * DIMH;
    float* mbuf  = p; p += NNODE * DIMH;
    float* gi    = p; p += NNODE * 192;
    float* gh    = p; p += NNODE * 192;
    float* ew    = p; p += NEDGE * DIMH;
    float* invd  = p; p += NNODE;
    float* qstar = p; p += NGRAPH * 128;       // qstar, hs, cs contiguous for one zero
    float* hs    = p; p += NGRAPH * DIMH;
    float* cs    = p; p += NGRAPH * DIMH;
    float* qb    = p; p += NGRAPH * DIMH;
    float* lstmo = p; p += NGRAPH * DIMH;
    float* z2    = p; p += NTOR * 320;
    float* z2h   = p; p += NTOR * DIMH;
    _Float16* bp_msg  = (_Float16*)p;
    _Float16* bp_root = bp_msg  + (size_t)KBMSG * 4 * 512;
    _Float16* bp_gih  = bp_root + 2 * 4 * 512;
    _Float16* bp_ghh  = bp_gih  + 2 * 12 * 512;
    _Float16* bp_lin1 = bp_ghh  + 2 * 12 * 512;

    // ---- setup ----
    k_lin0<<<cdiv(NNODE * DIMH, 256), 256, 0, stream>>>(x, lin0_w, lin0_b, h);
    k_enn1<<<cdiv(NEDGE * DIMH, 256), 256, 0, stream>>>(edge_attr, enn1_w, enn1_b, ew);
    k_zero<<<cdiv(NNODE, 256), 256, 0, stream>>>(invd, NNODE);
    k_deg<<<cdiv(NEDGE, 256), 256, 0, stream>>>(dst, invd);
    k_invdeg<<<cdiv(NNODE, 256), 256, 0, stream>>>(invd);

    k_pack_msg<<<cdiv(KBMSG * 4 * 512, 256), 256, 0, stream>>>(enn2_w, enn2_b, bp_msg);
    k_pack_G<<<cdiv(2 * 4 * 512, 256), 256, 0, stream>>>(root_w, bp_root, 64, 64);
    k_pack_T<<<cdiv(2 * 12 * 512, 256), 256, 0, stream>>>(gru_w_ih, bp_gih, 64, 192);
    k_pack_T<<<cdiv(2 * 12 * 512, 256), 256, 0, stream>>>(gru_w_hh, bp_ghh, 64, 192);
    k_pack_T<<<cdiv(10 * 4 * 512, 256), 256, 0, stream>>>(lin1_w, bp_lin1, 320, 64);

    // ---- 6 message-passing iterations ----
    for (int it = 0; it < 6; ++it) {
        k_zero<<<cdiv(NNODE * DIMH, 256), 256, 0, stream>>>(agg, NNODE * DIMH);
        k_msg_wmma<<<NEDGE / 16, 128, 0, stream>>>(h, ew, bp_msg, src, dst, agg);
        // mbuf = relu(out@root_w + conv_b + agg*invd)   (fused NNConv epilogue)
        k_gemm_wmma<<<dim3(NNODE / 16, 1), 128, 0, stream>>>(h, bp_root, conv_b, mbuf,
                                                             64, 64, 1, agg, invd);
        k_gemm_wmma<<<dim3(NNODE / 16, 3), 128, 0, stream>>>(mbuf, bp_gih, gru_b_ih, gi,
                                                             64, 192, 0, nullptr, nullptr);
        k_gemm_wmma<<<dim3(NNODE / 16, 3), 128, 0, stream>>>(h, bp_ghh, gru_b_hh, gh,
                                                             64, 192, 0, nullptr, nullptr);
        k_gru<<<cdiv(NNODE * DIMH, 256), 256, 0, stream>>>(gi, gh, h);
    }

    // ---- Set2Set (6 steps) ----
    k_zero<<<cdiv(NGRAPH * 256, 256), 256, 0, stream>>>(qstar, NGRAPH * 256); // qstar+hs+cs
    for (int s = 0; s < 6; ++s) {
        k_lstm<<<NGRAPH, 64, 0, stream>>>(qstar, 128, s2s_w_ih, s2s_b_ih,
                                          s2s_w_hh, s2s_b_hh, hs, cs, qb);
        k_attn<<<NGRAPH, 64, 0, stream>>>(h, qb, qstar);
    }

    // ---- memory LSTM (zero init state lives directly in d_out) ----
    k_zero<<<cdiv(NGRAPH * 128, 256), 256, 0, stream>>>(hx, NGRAPH * 128);    // hx+cx
    k_lstm<<<NGRAPH, 64, 0, stream>>>(qstar, 128, mem_w_ih, mem_b_ih,
                                      mem_w_hh, mem_b_hh, hx, cx, lstmo);

    // ---- head ----
    k_z2<<<cdiv(NTOR * 320, 256), 256, 0, stream>>>(lstmo, h, nonring, z2);
    k_gemm_wmma<<<dim3(NTOR / 16, 1), 128, 0, stream>>>(z2, bp_lin1, lin1_b, z2h,
                                                        320, 64, 1, nullptr, nullptr);
    k_head<<<cdiv(NTOR * 6, 256), 256, 0, stream>>>(z2h, lin2_w, lin2_b, out);
}
